// GCN_54065048323040
// MI455X (gfx1250) — matrix operand order repaired and checked by reading, not compiled, not run on hardware
//
#include <hip/hip_runtime.h>

// ---------------------------------------------------------------------------
// 3-layer GCN for MI455X (gfx1250, wave32).
//   per layer:  t = h @ W          (WMMA f16 split-precision, f32 accumulate)
//               agg = scatter_add(t[src]*norm -> dst)   (L2-resident HW f32 atomics)
//               h'  = relu?(agg + t*dinv^2 + b)
// ---------------------------------------------------------------------------

typedef __attribute__((ext_vector_type(16))) _Float16 v16h;
typedef __attribute__((ext_vector_type(8)))  _Float16 v8h;
typedef __attribute__((ext_vector_type(8)))  float    v8f;

#define LDS_PITCH 136   // 128 + 8 halfs: 272B row stride breaks 64-bank conflicts

// hardware f32 atomic add (GLOBAL_ATOMIC_ADD_F32), no CAS fallback
__device__ __forceinline__ void atom_add_f32(float* p, float v) {
  unsafeAtomicAdd(p, v);
}

// -------------------------------- utility kernels --------------------------

__global__ void gcn_fill(float* __restrict__ p, float v, int n) {
  int i = blockIdx.x * blockDim.x + threadIdx.x;
  if (i < n) p[i] = v;
}

__global__ void gcn_zero4(float4* __restrict__ p, int n4) {
  int i = blockIdx.x * blockDim.x + threadIdx.x;
  if (i < n4) p[i] = make_float4(0.f, 0.f, 0.f, 0.f);
}

__global__ void gcn_deg(const int* __restrict__ dst, float* __restrict__ deg, int E) {
  int e = blockIdx.x * blockDim.x + threadIdx.x;
  if (e < E) atom_add_f32(&deg[dst[e]], 1.0f);
}

__global__ void gcn_dinv(float* __restrict__ d, int n) {
  int i = blockIdx.x * blockDim.x + threadIdx.x;
  if (i < n) d[i] = rsqrtf(d[i]);
}

// -------- pack W[k][n] (f32) into per-lane WMMA B-fragment layout (hi+lo f16)
// Fragment address: ((nt*nchunks + c)*32 + lane)*16 + i
//   k = c*32 + (lane>>4)*16 + i ,  n = nt*16 + (lane&15)
__global__ void gcn_pack_w(const float* __restrict__ W,
                           _Float16* __restrict__ hi, _Float16* __restrict__ lo,
                           int Fin, int Fout) {
  int idx = blockIdx.x * blockDim.x + threadIdx.x;
  if (idx >= Fin * Fout) return;
  int k = idx / Fout, n = idx % Fout;
  int nchunks = Fin >> 5;
  int nt = n >> 4;
  int c  = k >> 5;
  int kr = k & 31;
  int l  = (kr & 16) | (n & 15);
  int i  = kr & 15;
  size_t p = (((size_t)nt * nchunks + c) * 32 + l) * 16 + i;
  float w = W[idx];
  _Float16 wh = (_Float16)w;
  hi[p] = wh;
  lo[p] = (_Float16)(w - (float)wh);
}

// -------------------------------- WMMA GEMM --------------------------------
// grid.x = N/16 (exact), blockDim = 32*(Fout/16); wave w computes tile (blk, w).
// Split precision: C += Ahi*Bhi + Alo*Bhi + Ahi*Blo  (f32 accumulate).
// FIN is a compile-time constant (128 for all layers) -> branch-free unroll.
template <int FIN>
__global__ void gcn_gemm_wmma(const float* __restrict__ A,
                              const _Float16* __restrict__ Bhi,
                              const _Float16* __restrict__ Blo,
                              float* __restrict__ C,
                              int Fout) {
  __shared__ _Float16 sAhi[16][LDS_PITCH];
  __shared__ _Float16 sAlo[16][LDS_PITCH];

  constexpr int NCHUNKS = FIN >> 5;
  const int lane = threadIdx.x & 31;
  const int wave = threadIdx.x >> 5;      // = ntile
  const int lh   = lane & 15;
  const int hi16 = lane >> 4;             // 0 or 1
  const int m0   = blockIdx.x * 16;

  // Stage the 16 x FIN f32 A-tile once per block, converting to hi/lo f16.
  for (int tt = threadIdx.x; tt < (16 * FIN) >> 2; tt += blockDim.x) {
    int r = (tt << 2) / FIN;
    int c = (tt << 2) % FIN;
    float4 v = *(const float4*)(A + (size_t)(m0 + r) * FIN + c);
    _Float16 h0 = (_Float16)v.x, h1 = (_Float16)v.y;
    _Float16 h2 = (_Float16)v.z, h3 = (_Float16)v.w;
    sAhi[r][c + 0] = h0; sAhi[r][c + 1] = h1;
    sAhi[r][c + 2] = h2; sAhi[r][c + 3] = h3;
    sAlo[r][c + 0] = (_Float16)(v.x - (float)h0);
    sAlo[r][c + 1] = (_Float16)(v.y - (float)h1);
    sAlo[r][c + 2] = (_Float16)(v.z - (float)h2);
    sAlo[r][c + 3] = (_Float16)(v.w - (float)h3);
  }
  __syncthreads();

  v8f acc = {};
  const _Float16* bph = Bhi + ((size_t)wave * NCHUNKS * 32 + lane) * 16;
  const _Float16* bpl = Blo + ((size_t)wave * NCHUNKS * 32 + lane) * 16;

  #pragma unroll
  for (int c = 0; c < NCHUNKS; ++c) {
    int ka = (c << 5) + hi16 * 16;       // A layout: lane<16 K[0..15], lane>=16 K[16..31]
    v8h a0 = *(const v8h*)&sAhi[lh][ka];
    v8h a1 = *(const v8h*)&sAhi[lh][ka + 8];
    v8h a2 = *(const v8h*)&sAlo[lh][ka];
    v8h a3 = *(const v8h*)&sAlo[lh][ka + 8];
    v16h ahi = __builtin_shufflevector(a0, a1, 0,1,2,3,4,5,6,7,8,9,10,11,12,13,14,15);
    v16h alo = __builtin_shufflevector(a2, a3, 0,1,2,3,4,5,6,7,8,9,10,11,12,13,14,15);
    v16h bhi = *(const v16h*)(bph + (size_t)c * 512);
    v16h blo = *(const v16h*)(bpl + (size_t)c * 512);
    acc = __builtin_amdgcn_wmma_f32_16x16x32_f16(false, ahi, false, bhi,
                                                 (short)0, acc, false, false);
    acc = __builtin_amdgcn_wmma_f32_16x16x32_f16(false, alo, false, bhi,
                                                 (short)0, acc, false, false);
    acc = __builtin_amdgcn_wmma_f32_16x16x32_f16(false, ahi, false, blo,
                                                 (short)0, acc, false, false);
  }

  // C/D layout: VGPR r -> row (r + 8*hi16), col lh (within the 16x16 tile).
  float* crow = C + (size_t)(m0 + 8 * hi16) * Fout + wave * 16 + lh;
  #pragma unroll
  for (int r = 0; r < 8; ++r) crow[(size_t)r * Fout] = acc[r];
}

// ------------------------------ edge scatter-add ---------------------------
// thread = (edge, float4-chunk); h/agg are L2-resident (51MB << 192MB L2).
__global__ void gcn_scatter(const float* __restrict__ t,
                            const int* __restrict__ src, const int* __restrict__ dst,
                            const float* __restrict__ dinv,
                            float* __restrict__ agg, int E, int F, int cshift) {
  int tid = blockIdx.x * blockDim.x + threadIdx.x;
  int e = tid >> cshift;
  if (e >= E) return;
  int c = (tid & ((1 << cshift) - 1)) << 2;
  int s = src[e], d = dst[e];
  float nrm = dinv[s] * dinv[d];
  float4 v = *(const float4*)(t + (size_t)s * F + c);
  float* a = agg + (size_t)d * F + c;
  atom_add_f32(a + 0, v.x * nrm);
  atom_add_f32(a + 1, v.y * nrm);
  atom_add_f32(a + 2, v.z * nrm);
  atom_add_f32(a + 3, v.w * nrm);
}

// ----------------------- epilogue: self-loop + bias + ReLU -----------------
__global__ void gcn_combine(const float* __restrict__ agg, const float* __restrict__ t,
                            const float* __restrict__ dinv, const float* __restrict__ bias,
                            float* __restrict__ out, int N, int F, int cshift, int relu) {
  int tid = blockIdx.x * blockDim.x + threadIdx.x;
  int row = tid >> cshift;
  if (row >= N) return;
  int c = (tid & ((1 << cshift) - 1)) << 2;
  float di = dinv[row];
  float sn = di * di;
  size_t off = (size_t)row * F + c;
  float4 av = *(const float4*)(agg + off);
  float4 tv = *(const float4*)(t + off);
  float4 bv = *(const float4*)(bias + c);
  float4 o;
  o.x = fmaf(tv.x, sn, av.x) + bv.x;
  o.y = fmaf(tv.y, sn, av.y) + bv.y;
  o.z = fmaf(tv.z, sn, av.z) + bv.z;
  o.w = fmaf(tv.w, sn, av.w) + bv.w;
  if (relu) {
    o.x = fmaxf(o.x, 0.f); o.y = fmaxf(o.y, 0.f);
    o.z = fmaxf(o.z, 0.f); o.w = fmaxf(o.w, 0.f);
  }
  *(float4*)(out + off) = o;
}

// --------------------------------- launch ----------------------------------

static inline int cdiv_i(long long a, long long b) { return (int)((a + b - 1) / b); }

extern "C" void kernel_launch(void* const* d_in, const int* in_sizes, int n_in,
                              void* d_out, int out_size, void* d_ws, size_t ws_size,
                              hipStream_t stream) {
  (void)n_in; (void)out_size; (void)ws_size;
  const int IN = 128, HD = 128, OC = 64;
  const float* x  = (const float*)d_in[0];
  const int*   ei = (const int*)d_in[1];
  const float* W1 = (const float*)d_in[2];
  const float* b1 = (const float*)d_in[3];
  const float* W2 = (const float*)d_in[4];
  const float* b2 = (const float*)d_in[5];
  const float* W3 = (const float*)d_in[6];
  const float* b3 = (const float*)d_in[7];

  const int N = in_sizes[0] / IN;     // 100000 (== 16 * 6250, boundary-free)
  const int E = in_sizes[1] / 2;      // 800000
  const int* src = ei;
  const int* dst = ei + E;

  // workspace layout (~154 MB)
  char* wsp = (char*)d_ws;
  auto carve = [&](size_t bytes) {
    char* p = wsp; wsp += (bytes + 255) & ~(size_t)255; return (void*)p;
  };
  float*    dinv = (float*)carve((size_t)N * 4);
  float*    t    = (float*)carve((size_t)N * HD * 4);
  float*    agg  = (float*)carve((size_t)N * HD * 4);
  float*    h    = (float*)carve((size_t)N * HD * 4);
  _Float16* wph  = (_Float16*)carve((size_t)IN * HD * 2);
  _Float16* wpl  = (_Float16*)carve((size_t)IN * HD * 2);

  const int mtiles = N / 16;

  // degrees -> dinv
  gcn_fill<<<cdiv_i(N, 256), 256, 0, stream>>>(dinv, 1.0f, N);
  gcn_deg <<<cdiv_i(E, 256), 256, 0, stream>>>(dst, dinv, E);
  gcn_dinv<<<cdiv_i(N, 256), 256, 0, stream>>>(dinv, N);

  // ---- layer 1: x[IN] -> h[HD], ReLU
  gcn_pack_w<<<cdiv_i((long long)IN * HD, 256), 256, 0, stream>>>(W1, wph, wpl, IN, HD);
  gcn_gemm_wmma<128><<<mtiles, 32 * (HD / 16), 0, stream>>>(x, wph, wpl, t, HD);
  gcn_zero4<<<cdiv_i((long long)N * HD / 4, 256), 256, 0, stream>>>((float4*)agg, N * (HD / 4));
  gcn_scatter<<<cdiv_i((long long)E * (HD / 4), 256), 256, 0, stream>>>(t, src, dst, dinv, agg, E, HD, 5);
  gcn_combine<<<cdiv_i((long long)N * (HD / 4), 256), 256, 0, stream>>>(agg, t, dinv, b1, h, N, HD, 5, 1);

  // ---- layer 2: h[HD] -> h[HD], ReLU
  gcn_pack_w<<<cdiv_i((long long)HD * HD, 256), 256, 0, stream>>>(W2, wph, wpl, HD, HD);
  gcn_gemm_wmma<128><<<mtiles, 32 * (HD / 16), 0, stream>>>(h, wph, wpl, t, HD);
  gcn_zero4<<<cdiv_i((long long)N * HD / 4, 256), 256, 0, stream>>>((float4*)agg, N * (HD / 4));
  gcn_scatter<<<cdiv_i((long long)E * (HD / 4), 256), 256, 0, stream>>>(t, src, dst, dinv, agg, E, HD, 5);
  gcn_combine<<<cdiv_i((long long)N * (HD / 4), 256), 256, 0, stream>>>(agg, t, dinv, b2, h, N, HD, 5, 1);

  // ---- layer 3: h[HD] -> out[OC], no ReLU
  gcn_pack_w<<<cdiv_i((long long)HD * OC, 256), 256, 0, stream>>>(W3, wph, wpl, HD, OC);
  gcn_gemm_wmma<128><<<mtiles, 32 * (OC / 16), 0, stream>>>(h, wph, wpl, t, OC);
  gcn_zero4<<<cdiv_i((long long)N * OC / 4, 256), 256, 0, stream>>>((float4*)agg, N * (OC / 4));
  gcn_scatter<<<cdiv_i((long long)E * (OC / 4), 256), 256, 0, stream>>>(t, src, dst, dinv, agg, E, OC, 4);
  gcn_combine<<<cdiv_i((long long)N * (OC / 4), 256), 256, 0, stream>>>(agg, t, dinv, b3, (float*)d_out, N, OC, 4, 0);
}
